// GCNConv_23433341567794
// MI455X (gfx1250) — matrix operand order repaired and checked by reading, not compiled, not run on hardware
//
#include <hip/hip_runtime.h>
#include <hip/hip_bf16.h>

// ---------------------------------------------------------------------------
// GCNConv for MI455X (gfx1250, wave32, WMMA + TDM):
//   Phase 1: Xp = X @ W    -> v_wmma_f32_16x16x32_bf16, B-matrix staged into
//                             LDS via tensor_load_to_lds (TDM), A from HBM
//   Phase 2: out = A_csr @ Xp -> wave-per-row gather/sum (L2-resident Xp)
// ---------------------------------------------------------------------------

typedef __attribute__((ext_vector_type(16))) __bf16       v16bf;
typedef __attribute__((ext_vector_type(8)))  float        v8f;
typedef __attribute__((ext_vector_type(4)))  unsigned int v4u;
typedef __attribute__((ext_vector_type(8)))  int          v8i;
typedef __attribute__((ext_vector_type(4)))  int          v4i;

#define D_IN_  256
#define D_OUT_ 128

// ---------------------------------------------------------------------------
// Kernel 1: W [K=256][N=128] f32 row-major  ->  Wt [N=128][K=256] bf16.
// Pre-transposing makes every WMMA B-fragment a contiguous 16-element (32 B)
// run per lane, in the same contiguous k order A uses (not the ISA table order):
//   lanes 0-15 : N = lane,    K = k0 .. k0+15   (2 bf16 packed per VGPR)
//   lanes 16-31: N = lane-16, K = k0+16 .. k0+31
// ---------------------------------------------------------------------------
__global__ void wt_convert_kernel(const float* __restrict__ W,
                                  __bf16* __restrict__ Wt) {
    int idx = blockIdx.x * blockDim.x + threadIdx.x;   // 0 .. 32767
    if (idx >= D_IN_ * D_OUT_) return;
    int k = idx >> 7;          // 0..255
    int n = idx & 127;         // 0..127
    Wt[(size_t)n * D_IN_ + k] = (__bf16)W[(size_t)k * D_OUT_ + n];
}

// ---------------------------------------------------------------------------
// Kernel 2: Xp = X @ W via v_wmma_f32_16x16x32_bf16.
// Block = 256 threads = 8 waves; block tile = 128 rows x 128 cols.
// Wt (64 KB bf16) is DMA'd into LDS once per workgroup by the Tensor Data
// Mover. Inner loop shape (enforced by an asm register pin):
//   issue next-iter A loads -> issue 16 ds_load_b128 -> cvt A -> one
//   s_wait_dscnt -> 8 WMMAs back-to-back into independent accumulators.
// ---------------------------------------------------------------------------
__global__ void __launch_bounds__(256)
gemm_xw_wmma_kernel(const float* __restrict__ X,
                    const __bf16* __restrict__ Wt,
                    float* __restrict__ Xp,
                    int nNodes) {
    __shared__ __bf16 ldsW[D_OUT_ * D_IN_];   // 128 x 256 bf16 = 64 KB

    const int wave = threadIdx.x >> 5;     // 0..7
    const int lane = threadIdx.x & 31;
    const int half = lane >> 4;            // which K half this lane carries
    const int m    = lane & 15;            // row within the 16-row wave tile
    const int rowBase = blockIdx.x * 128 + wave * 16;

    // ---- TDM: stage Wt[128][256] bf16 (64 KB) into LDS (wave 0 issues) ----
    if (wave == 0) {
        unsigned long long ga = (unsigned long long)(const void*)Wt;
        // D# group 0 (ISA 8.3): count=1 | lds_addr=0 | global_addr | type=2
        v4u g0;
        g0[0] = 1u;                                   // count=1, user-mode
        g0[1] = 0u;                                   // lds_addr = 0 (bytes)
        g0[2] = (unsigned)(ga & 0xFFFFFFFFu);         // global_addr[31:0]
        g0[3] = (unsigned)((ga >> 32) & 0x01FFFFFFu)  // global_addr[56:32]
              | 0x80000000u;                          // type = 2 ("image")
        // D# group 1 (ISA 8.4): data_size=2B, tensor 256x128, tile 256x128,
        // dim0 stride = 256 elements.
        v8i g1;
        g1[0] = (int)(1u << 16);            // wg_mask=0, data_size=1 (2 bytes)
        g1[1] = (int)(256u << 16);          // tensor_dim0[15:0] @ bits 63:48
        g1[2] = (int)((256u >> 16) | (128u << 16)); // dim0 hi | dim1 lo
        g1[3] = (int)((128u >> 16) | (256u << 16)); // dim1 hi | tile_dim0=256
        g1[4] = (int)128u;                  // tile_dim1=128, tile_dim2=0
        g1[5] = (int)256u;                  // tensor_dim0_stride[31:0] = 256
        g1[6] = 0;
        g1[7] = 0;
        v4i gz = {0, 0, 0, 0};
#if defined(__clang_major__) && (__clang_major__ >= 23)
        v8i gz8 = {0, 0, 0, 0, 0, 0, 0, 0};
        __builtin_amdgcn_tensor_load_to_lds(g0, g1, gz, gz, gz8, 0);
#else
        __builtin_amdgcn_tensor_load_to_lds(g0, g1, gz, gz, 0);
#endif
        __builtin_amdgcn_s_wait_tensorcnt(0);
    }
    __syncthreads();

    // The TDM wrote LDS behind the compiler's back: escape the LDS base and
    // clobber memory so the ds_loads below cannot be folded to undef.
    {
        unsigned ldsEsc = (unsigned)(unsigned long long)(void*)&ldsW[0];
        asm volatile("" : : "v"(ldsEsc) : "memory");
    }

    // Clamp A-row for loads (WMMA requires EXEC all-ones; garbage rows are
    // simply never stored).
    int rowA = rowBase + m;
    if (rowA >= nNodes) rowA = nNodes - 1;
    const float* __restrict__ arow = X + (size_t)rowA * D_IN_;

    v8f acc[8];
#pragma unroll
    for (int t = 0; t < 8; ++t) acc[t] = v8f{};

    const __bf16* wkBase = ldsW + (size_t)m * D_IN_ + half * 16;

    // Prologue: A raw loads for k0 = 0.
    float4 f0 = *(const float4*)(arow + half * 16);
    float4 f1 = *(const float4*)(arow + half * 16 + 4);
    float4 f2 = *(const float4*)(arow + half * 16 + 8);
    float4 f3 = *(const float4*)(arow + half * 16 + 12);

#pragma unroll
    for (int kk = 0; kk < 8; ++kk) {
        const int k0 = kk * 32;

        // ---- software pipeline: issue next iteration's A loads early ------
        float4 n0, n1, n2, n3;
        if (kk < 7) {
            const int kn = k0 + 32;
            n0 = *(const float4*)(arow + kn + half * 16);
            n1 = *(const float4*)(arow + kn + half * 16 + 4);
            n2 = *(const float4*)(arow + kn + half * 16 + 8);
            n3 = *(const float4*)(arow + kn + half * 16 + 12);
        }

        // ---- issue all 16 ds_load_b128 for the 8 B fragments --------------
        const __bf16* wk = wkBase + k0;
        v16bf b0 = *(const v16bf*)(wk + 0 * 16 * D_IN_);
        v16bf b1 = *(const v16bf*)(wk + 1 * 16 * D_IN_);
        v16bf b2 = *(const v16bf*)(wk + 2 * 16 * D_IN_);
        v16bf b3 = *(const v16bf*)(wk + 3 * 16 * D_IN_);
        v16bf b4 = *(const v16bf*)(wk + 4 * 16 * D_IN_);
        v16bf b5 = *(const v16bf*)(wk + 5 * 16 * D_IN_);
        v16bf b6 = *(const v16bf*)(wk + 6 * 16 * D_IN_);
        v16bf b7 = *(const v16bf*)(wk + 7 * 16 * D_IN_);

        // ---- convert current A: one contiguous run of 16 f32 -> 16 bf16 ---
        v16bf a;
        a[0]  = (__bf16)f0.x; a[1]  = (__bf16)f0.y;
        a[2]  = (__bf16)f0.z; a[3]  = (__bf16)f0.w;
        a[4]  = (__bf16)f1.x; a[5]  = (__bf16)f1.y;
        a[6]  = (__bf16)f1.z; a[7]  = (__bf16)f1.w;
        a[8]  = (__bf16)f2.x; a[9]  = (__bf16)f2.y;
        a[10] = (__bf16)f2.z; a[11] = (__bf16)f2.w;
        a[12] = (__bf16)f3.x; a[13] = (__bf16)f3.y;
        a[14] = (__bf16)f3.z; a[15] = (__bf16)f3.w;

        // ---- pin fragments in distinct registers: forces all ds_loads to
        //      issue (and a single dscnt wait) before the WMMA burst --------
        asm volatile("" : "+v"(b0), "+v"(b1), "+v"(b2), "+v"(b3),
                          "+v"(b4), "+v"(b5), "+v"(b6), "+v"(b7), "+v"(a));

        // ---- 8 WMMAs back-to-back (independent accumulators) --------------
        acc[0] = __builtin_amdgcn_wmma_f32_16x16x32_bf16(false, a, false, b0,
                                                         (short)0, acc[0], false, false);
        acc[1] = __builtin_amdgcn_wmma_f32_16x16x32_bf16(false, a, false, b1,
                                                         (short)0, acc[1], false, false);
        acc[2] = __builtin_amdgcn_wmma_f32_16x16x32_bf16(false, a, false, b2,
                                                         (short)0, acc[2], false, false);
        acc[3] = __builtin_amdgcn_wmma_f32_16x16x32_bf16(false, a, false, b3,
                                                         (short)0, acc[3], false, false);
        acc[4] = __builtin_amdgcn_wmma_f32_16x16x32_bf16(false, a, false, b4,
                                                         (short)0, acc[4], false, false);
        acc[5] = __builtin_amdgcn_wmma_f32_16x16x32_bf16(false, a, false, b5,
                                                         (short)0, acc[5], false, false);
        acc[6] = __builtin_amdgcn_wmma_f32_16x16x32_bf16(false, a, false, b6,
                                                         (short)0, acc[6], false, false);
        acc[7] = __builtin_amdgcn_wmma_f32_16x16x32_bf16(false, a, false, b7,
                                                         (short)0, acc[7], false, false);

        f0 = n0; f1 = n1; f2 = n2; f3 = n3;
    }

    // ---- store C/D (16x16 f32 layout: VGPR r -> M=r (lanes 0-15) / M=r+8) --
    const int mStoreBase = rowBase + half * 8;
#pragma unroll
    for (int t = 0; t < 8; ++t) {
#pragma unroll
        for (int r = 0; r < 8; ++r) {
            int row = mStoreBase + r;
            if (row < nNodes)
                Xp[(size_t)row * D_OUT_ + t * 16 + m] = acc[t][r];
        }
    }
}

// ---------------------------------------------------------------------------
// Kernel 3: CSR aggregation  out[i] = sum_{j in [rp[i],rp[i+1])} Xp[col[j]].
// One wave per node row; lane owns one float4 (4 cols) -> a neighbor row is
// one coalesced 512 B burst. Xp (25.6 MB) is L2-resident on MI455X (192 MB),
// so the irregular gather runs at L2 bandwidth.
// ---------------------------------------------------------------------------
__global__ void __launch_bounds__(256)
csr_aggregate_kernel(const float* __restrict__ Xp,
                     const int* __restrict__ row_pointers,
                     const int* __restrict__ column_index,
                     float* __restrict__ out,
                     int nNodes) {
    const int row  = blockIdx.x * 8 + (threadIdx.x >> 5);
    const int lane = threadIdx.x & 31;
    if (row >= nNodes) return;

    const int s = row_pointers[row];
    const int e = row_pointers[row + 1];

    const float4* __restrict__ src = (const float4*)Xp;   // 32 float4 per row
    float4 acc = {0.f, 0.f, 0.f, 0.f};

    int j = s;
    // 2-way unroll to keep two gathers in flight per lane.
    for (; j + 1 < e; j += 2) {
        int c0 = column_index[j];
        int c1 = column_index[j + 1];
        float4 v0 = src[(size_t)c0 * 32 + lane];
        float4 v1 = src[(size_t)c1 * 32 + lane];
        acc.x += v0.x + v1.x;
        acc.y += v0.y + v1.y;
        acc.z += v0.z + v1.z;
        acc.w += v0.w + v1.w;
    }
    for (; j < e; ++j) {
        int c = column_index[j];
        float4 v = src[(size_t)c * 32 + lane];
        acc.x += v.x; acc.y += v.y; acc.z += v.z; acc.w += v.w;
    }

    ((float4*)out)[(size_t)row * 32 + lane] = acc;
}

// ---------------------------------------------------------------------------
// Host-side launcher (graph-capture safe: only kernel launches on `stream`).
// Inputs (setup_inputs order):
//   0: X [N,256] f32   1: W [256,128] f32   2: row_pointers [N+1] i32
//   3: column_index [E] i32   4..9: unused tiling metadata
// Workspace: [ Wt bf16 64 KB | Xp f32 N*128*4 B ]
// ---------------------------------------------------------------------------
extern "C" void kernel_launch(void* const* d_in, const int* in_sizes, int n_in,
                              void* d_out, int out_size, void* d_ws, size_t ws_size,
                              hipStream_t stream) {
    const float* X      = (const float*)d_in[0];
    const float* W      = (const float*)d_in[1];
    const int*   rowptr = (const int*)d_in[2];
    const int*   colidx = (const int*)d_in[3];
    float*       out    = (float*)d_out;

    const int nNodes = in_sizes[2] - 1;          // 50000

    __bf16* Wt = (__bf16*)d_ws;                                  // 64 KB
    float*  Xp = (float*)((char*)d_ws + (size_t)D_IN_ * D_OUT_ * sizeof(__bf16));

    // 1) transpose+convert W -> Wt (bf16, [N][K])
    {
        int total  = D_IN_ * D_OUT_;             // 32768
        int blocks = (total + 255) / 256;
        wt_convert_kernel<<<blocks, 256, 0, stream>>>(W, Wt);
    }
    // 2) Xp = X @ W via WMMA (TDM stages Wt into LDS per workgroup)
    {
        int blocks = (nNodes + 127) / 128;       // 391
        gemm_xw_wmma_kernel<<<blocks, 256, 0, stream>>>(X, Wt, Xp, nNodes);
    }
    // 3) out = A_csr @ Xp
    {
        int blocks = (nNodes + 7) / 8;           // 6250
        csr_aggregate_kernel<<<blocks, 256, 0, stream>>>(Xp, rowptr, colidx,
                                                         out, nNodes);
    }
}
